// NeuroDynMoE_51857435132579
// MI455X (gfx1250) — matrix operand order repaired and checked by reading, compiled, not run on hardware
//
#include <hip/hip_runtime.h>

// ---------------------------------------------------------------------------
// NeuroDynMoE for MI455X (gfx1250, wave32, WMMA bf16 16x16x32)
//   B=16384 D=256 E=8 H=128 O=1 TOPK=2
// Routed MoE: gate+top2 -> per-expert row lists -> dense bf16 WMMA GEMMs
// over gathered 128-row tiles, everything staged in LDS (~200KB of the
// 320KB/WGP), core kept as bf16 in LDS between the two GEMMs.
// ---------------------------------------------------------------------------

#define B_ROWS 16384
#define D_DIM  256
#define E_EXP  8
#define H_DIM  128

typedef __attribute__((ext_vector_type(16))) __bf16          v16bf;
typedef __attribute__((ext_vector_type(8)))  float           v8f;
typedef __attribute__((ext_vector_type(8)))  unsigned short  ushort8;

union Frag { v16bf v; ushort8 h[2]; };

static __device__ __forceinline__ unsigned short f2bf(float f) {
    unsigned u = __builtin_bit_cast(unsigned, f);
    u += 0x7FFFu + ((u >> 16) & 1u);          // round-to-nearest-even
    return (unsigned short)(u >> 16);
}
static __device__ __forceinline__ float bf2f(unsigned short s) {
    unsigned u = ((unsigned)s) << 16;
    return __builtin_bit_cast(float, u);
}
static __device__ __forceinline__ v8f vzero8() {
    v8f z;
#pragma unroll
    for (int j = 0; j < 8; ++j) z[j] = 0.0f;
    return z;
}

// ---------------------------------------------------------------------------
// Kernel 0: transpose + convert weights to bf16.
//   W1T[e][h][d] = bf16(W1[e][d][h])   (E*H*D = 262144)
//   WpT[e][n][k] = bf16(Wp[e][k][n])   (E*H*H = 131072)
// ---------------------------------------------------------------------------
__global__ void prep_kernel(const float* __restrict__ W1,
                            const float* __restrict__ Wp,
                            unsigned short* __restrict__ W1T,
                            unsigned short* __restrict__ WpT) {
    const int i  = blockIdx.x * 256 + threadIdx.x;
    const int n1 = E_EXP * H_DIM * D_DIM;
    if (i < n1) {
        int e = i / (H_DIM * D_DIM);
        int r = i - e * H_DIM * D_DIM;
        int h = r / D_DIM;
        int d = r - h * D_DIM;
        W1T[i] = f2bf(W1[(e * D_DIM + d) * H_DIM + h]);
    } else {
        int j = i - n1;
        if (j < E_EXP * H_DIM * H_DIM) {
            int e = j / (H_DIM * H_DIM);
            int r = j - e * H_DIM * H_DIM;
            int n = r / H_DIM;
            int k = r - n * H_DIM;
            WpT[j] = f2bf(Wp[(e * H_DIM + k) * H_DIM + n]);
        }
    }
}

// ---------------------------------------------------------------------------
// Kernel 1: gating. One thread per row: logits = f @ Wg, softmax, top-2,
// renormalize, append (row, weight) to the two experts' dispatch lists.
// ---------------------------------------------------------------------------
__global__ void gate_kernel(const float* __restrict__ feat,
                            const float* __restrict__ Wg,
                            int*   __restrict__ counts,
                            int*   __restrict__ rowlist,
                            float* __restrict__ roww) {
    __shared__ float WgS[D_DIM * E_EXP];            // 8 KB
    const int tid = threadIdx.x;
    for (int i = tid; i < D_DIM * E_EXP; i += 256) WgS[i] = Wg[i];
    __syncthreads();

    const int row = blockIdx.x * 256 + tid;
    float lg[E_EXP];
#pragma unroll
    for (int e = 0; e < E_EXP; ++e) lg[e] = 0.0f;

    const float4* f4 = (const float4*)(feat + (size_t)row * D_DIM);
    for (int d4 = 0; d4 < D_DIM / 4; ++d4) {
        float4 f = f4[d4];
        int d = d4 * 4;
#pragma unroll
        for (int e = 0; e < E_EXP; ++e) {
            lg[e] += f.x * WgS[(d + 0) * E_EXP + e] + f.y * WgS[(d + 1) * E_EXP + e]
                   + f.z * WgS[(d + 2) * E_EXP + e] + f.w * WgS[(d + 3) * E_EXP + e];
        }
    }
    float mx = lg[0];
#pragma unroll
    for (int e = 1; e < E_EXP; ++e) mx = fmaxf(mx, lg[e]);
    float g[E_EXP], s = 0.0f;
#pragma unroll
    for (int e = 0; e < E_EXP; ++e) { g[e] = expf(lg[e] - mx); s += g[e]; }
    float inv = 1.0f / s;
#pragma unroll
    for (int e = 0; e < E_EXP; ++e) g[e] *= inv;

    int i0 = 0;
#pragma unroll
    for (int e = 1; e < E_EXP; ++e) if (g[e] > g[i0]) i0 = e;
    int i1 = (i0 == 0) ? 1 : 0;
#pragma unroll
    for (int e = 0; e < E_EXP; ++e) if (e != i0 && g[e] > g[i1]) i1 = e;

    float w0 = g[i0], w1 = g[i1];
    float sw = w0 + w1 + 1e-10f;
    w0 /= sw; w1 /= sw;

    int s0 = atomicAdd(&counts[i0], 1);
    rowlist[i0 * B_ROWS + s0] = row;  roww[i0 * B_ROWS + s0] = w0;
    int s1 = atomicAdd(&counts[i1], 1);
    rowlist[i1 * B_ROWS + s1] = row;  roww[i1 * B_ROWS + s1] = w1;
}

// ---------------------------------------------------------------------------
// Kernel 2: expert forward on gathered 128-row tiles.
// LDS layout (ushort units):
//   Xs : 128 x 264  (features tile, bf16, padded)        [0      .. 33792)
//   W1s: 128 x 264  (W1T[e], [H][D] bf16, padded)        [33792  .. 67584)
//   Wps: 128 x 136  (WpT[e], [H][H] bf16, padded)        [67584  .. 84992)
//   Cs : 128 x 136  (relu core, bf16, padded)            [84992  .. 102400)
// total 102400 ushorts = 204800 bytes dynamic LDS.
// ---------------------------------------------------------------------------
#define XS_OFF 0
#define W1_OFF 33792
#define WP_OFF 67584
#define CS_OFF 84992
#define XS_STR 264
#define CS_STR 136
#define SMEM_USHORTS 102400

__global__ __launch_bounds__(256) void expert_kernel(
        const float* __restrict__ feat,
        const float* __restrict__ b1,
        const float* __restrict__ bpv,
        const float* __restrict__ mix_logit,
        const float* __restrict__ Wo,
        const float* __restrict__ bo,
        const unsigned short* __restrict__ W1T,
        const unsigned short* __restrict__ WpT,
        const int*   __restrict__ counts,
        const int*   __restrict__ rowlist,
        const float* __restrict__ roww,
        float* __restrict__ out) {
    extern __shared__ unsigned short sm[];

    const int e        = blockIdx.y;
    const int nrows    = counts[e];
    const int row_base = blockIdx.x * 128;
    if (row_base >= nrows) return;

    const int tid = threadIdx.x;

    // ---- stage: gather feature rows -> Xs (f32 -> bf16) -------------------
    for (int v = tid; v < 128 * 64; v += 256) {       // 64 float4 per row
        int r  = v >> 6;
        int c4 = (v & 63) << 2;
        int gi = row_base + r;
        float4 f = make_float4(0.f, 0.f, 0.f, 0.f);
        if (gi < nrows) {
            int grow = rowlist[e * B_ROWS + gi];
            f = *(const float4*)(feat + (size_t)grow * D_DIM + c4);
        }
        unsigned lo = (unsigned)f2bf(f.x) | ((unsigned)f2bf(f.y) << 16);
        unsigned hi = (unsigned)f2bf(f.z) | ((unsigned)f2bf(f.w) << 16);
        uint2 pk; pk.x = lo; pk.y = hi;
        *(uint2*)(sm + XS_OFF + r * XS_STR + c4) = pk;
    }
    // ---- stage: W1T[e] -> W1s --------------------------------------------
    {
        const ushort8* g = (const ushort8*)(W1T + (size_t)e * H_DIM * D_DIM);
        for (int v = tid; v < 128 * 32; v += 256) {   // 32 ushort8 per row
            int r = v >> 5, c8 = (v & 31) << 3;
            *(ushort8*)(sm + W1_OFF + r * XS_STR + c8) = g[v];
        }
    }
    // ---- stage: WpT[e] -> Wps --------------------------------------------
    {
        const ushort8* g = (const ushort8*)(WpT + (size_t)e * H_DIM * H_DIM);
        for (int v = tid; v < 128 * 16; v += 256) {   // 16 ushort8 per row
            int r = v >> 4, c8 = (v & 15) << 3;
            *(ushort8*)(sm + WP_OFF + r * CS_STR + c8) = g[v];
        }
    }
    __syncthreads();

    const int wave = tid >> 5;
    const int lane = tid & 31;
    const int ln   = lane & 15;
    const int kg   = lane >> 4;

    // ---- GEMM1: core = relu(X @ W1 + b1) ---------------------------------
    v8f acc[8];
#pragma unroll
    for (int i = 0; i < 8; ++i) acc[i] = vzero8();

    const unsigned short* xrow = sm + XS_OFF + (16 * wave + ln) * XS_STR;
#pragma unroll
    for (int kb = 0; kb < 8; ++kb) {                  // K = 256 / 32
        Frag a;
        const unsigned short* ap = xrow + kb * 32 + 8 * kg;
        a.h[0] = *(const ushort8*)(ap);
        a.h[1] = *(const ushort8*)(ap + 16);
#pragma unroll
        for (int nb = 0; nb < 8; ++nb) {              // N = 128 / 16
            Frag b;
            const unsigned short* bq = sm + W1_OFF + (nb * 16 + ln) * XS_STR + kb * 32 + 8 * kg;
            b.h[0] = *(const ushort8*)(bq);
            b.h[1] = *(const ushort8*)(bq + 16);
            acc[nb] = __builtin_amdgcn_wmma_f32_16x16x32_bf16(
                false, a.v, false, b.v, (short)0, acc[nb], false, false);
        }
    }

    const float* b1e = b1 + e * H_DIM;
#pragma unroll
    for (int nb = 0; nb < 8; ++nb) {
        int n = nb * 16 + ln;
        float bias = b1e[n];
#pragma unroll
        for (int r = 0; r < 8; ++r) {
            float c = acc[nb][r] + bias;
            c = c > 0.f ? c : 0.f;                    // relu
            sm[CS_OFF + (16 * wave + r + 8 * kg) * CS_STR + n] = f2bf(c);
        }
    }
    __syncthreads();

    // ---- GEMM2: plast_pre = core @ Wp ------------------------------------
    v8f acc2[8];
#pragma unroll
    for (int i = 0; i < 8; ++i) acc2[i] = vzero8();

    const unsigned short* crow = sm + CS_OFF + (16 * wave + ln) * CS_STR;
#pragma unroll
    for (int kb = 0; kb < 4; ++kb) {                  // K = 128 / 32
        Frag a;
        const unsigned short* ap = crow + kb * 32 + 8 * kg;
        a.h[0] = *(const ushort8*)(ap);
        a.h[1] = *(const ushort8*)(ap + 16);
#pragma unroll
        for (int nb = 0; nb < 8; ++nb) {
            Frag b;
            const unsigned short* bq = sm + WP_OFF + (nb * 16 + ln) * CS_STR + kb * 32 + 8 * kg;
            b.h[0] = *(const ushort8*)(bq);
            b.h[1] = *(const ushort8*)(bq + 16);
            acc2[nb] = __builtin_amdgcn_wmma_f32_16x16x32_bf16(
                false, a.v, false, b.v, (short)0, acc2[nb], false, false);
        }
    }

    // ---- epilogue: tanh, mix, dot with Wo, reduce, scatter ---------------
    const float mixv = 1.0f / (1.0f + expf(-mix_logit[e]));
    const float* bpe = bpv + e * H_DIM;
    const float* Woe = Wo + e * H_DIM;                // O == 1
    float pacc[8];
#pragma unroll
    for (int r = 0; r < 8; ++r) pacc[r] = 0.0f;

#pragma unroll
    for (int nb = 0; nb < 8; ++nb) {
        int n = nb * 16 + ln;
        float biasp = bpe[n];
        float wo    = Woe[n];
#pragma unroll
        for (int r = 0; r < 8; ++r) {
            int m = 16 * wave + r + 8 * kg;
            float core  = bf2f(sm[CS_OFF + m * CS_STR + n]);
            float pl    = tanhf(acc2[nb][r] + biasp);
            float mixed = core * (1.0f - mixv) + pl * mixv;
            pacc[r] += mixed * wo;
        }
    }
    // reduce the 16 column-lanes within each half-wave (rows differ by 8)
#pragma unroll
    for (int off = 8; off >= 1; off >>= 1) {
#pragma unroll
        for (int r = 0; r < 8; ++r) pacc[r] += __shfl_xor(pacc[r], off, 16);
    }
    if (ln == 0) {
        float boe = bo[e];
#pragma unroll
        for (int r = 0; r < 8; ++r) {
            int mloc = 16 * wave + r + 8 * kg;
            int gi   = row_base + mloc;
            if (gi < nrows) {
                int   grow = rowlist[e * B_ROWS + gi];
                float wgt  = roww[e * B_ROWS + gi];
                atomicAdd(out + grow, wgt * (pacc[r] + boe));
            }
        }
    }
}

// ---------------------------------------------------------------------------
// Workspace layout (bytes):
//   [0, 32)            counts[E]           (reserve 256)
//   [256, +512K)       rowlist[E][B] int
//   [524544, +512K)    roww[E][B] float
//   [1048832, +512K)   W1T bf16
//   [1573120, +256K)   WpT bf16             total ~1.75 MB
// ---------------------------------------------------------------------------
extern "C" void kernel_launch(void* const* d_in, const int* in_sizes, int n_in,
                              void* d_out, int out_size, void* d_ws, size_t ws_size,
                              hipStream_t stream) {
    const float* feat = (const float*)d_in[0];
    const float* Wg   = (const float*)d_in[1];
    const float* W1   = (const float*)d_in[2];
    const float* b1   = (const float*)d_in[3];
    const float* Wp   = (const float*)d_in[4];
    const float* bpv  = (const float*)d_in[5];
    const float* mixl = (const float*)d_in[6];
    const float* Wo   = (const float*)d_in[7];
    const float* bo   = (const float*)d_in[8];
    float* out = (float*)d_out;

    char* ws = (char*)d_ws;
    int*            counts  = (int*)(ws + 0);
    int*            rowlist = (int*)(ws + 256);
    float*          roww    = (float*)(ws + 524544);
    unsigned short* W1T     = (unsigned short*)(ws + 1048832);
    unsigned short* WpT     = (unsigned short*)(ws + 1573120);

    hipMemsetAsync(counts, 0, 256, stream);
    hipMemsetAsync(d_out, 0, (size_t)out_size * sizeof(float), stream);

    prep_kernel<<<1536, 256, 0, stream>>>(W1, Wp, W1T, WpT);
    gate_kernel<<<B_ROWS / 256, 256, 0, stream>>>(feat, Wg, counts, rowlist, roww);

    dim3 grid(B_ROWS / 128, E_EXP);
    size_t smem = (size_t)SMEM_USHORTS * sizeof(unsigned short);   // 204800 B
    expert_kernel<<<grid, 256, smem, stream>>>(feat, b1, bpv, mixl, Wo, bo,
                                               W1T, WpT, counts, rowlist, roww, out);
}
